// GATNet_40699110097099
// MI455X (gfx1250) — matrix operand order repaired and checked by reading, not compiled, not run on hardware
//
#include <hip/hip_runtime.h>
#include <hip/hip_bf16.h>
#include <math.h>

// ---------------------------------------------------------------------------
// GATNet for MI455X (gfx1250): WMMA bf16 GEMMs (async-to-LDS staging, 4 WMMA
// per wave per K-step with reuse-A hint) + graph attention scatter ops
// ---------------------------------------------------------------------------

typedef __bf16 v16bf __attribute__((ext_vector_type(16)));
typedef __bf16 v8bf  __attribute__((ext_vector_type(8)));
typedef float  v8f   __attribute__((ext_vector_type(8)));

#define NEG_SLOPE 0.2f

__device__ __forceinline__ __bf16 f2bf(float f) {
  // round-to-nearest-even fp32 -> bf16
  unsigned u = __float_as_uint(f);
  unsigned r = (u + 0x7FFFu + ((u >> 16) & 1u)) >> 16;
  unsigned short h = (unsigned short)r;
  __bf16 out;
  __builtin_memcpy(&out, &h, sizeof(out));
  return out;
}

__device__ __forceinline__ void atomicMaxF(float* addr, float val) {
  if (val >= 0.0f) atomicMax((int*)addr, __float_as_int(val));
  else             atomicMin((unsigned int*)addr, __float_as_uint(val));
}

// CDNA5 async global->LDS copy (8 bytes), tracked with ASYNCcnt.
__device__ __forceinline__ void async_copy_b64(__bf16* lds_dst, const __bf16* gsrc) {
  unsigned loff = (unsigned)(uintptr_t)lds_dst;   // LDS byte offset (low bits)
  asm volatile("global_load_async_to_lds_b64 %0, %1, off"
               :: "v"(loff), "v"(gsrc)
               : "memory");
}
__device__ __forceinline__ void wait_async0() {
  asm volatile("s_wait_asynccnt 0x0" ::: "memory");
}

// ---------------------------------------------------------------------------
// WMMA bf16 GEMM: C[M,N] = A[M,K] @ B[K,N]; A,B pre-converted bf16, K%32==0,
// M%64==0, N%128==0 (guaranteed by padding/sizes).
// Block = 256 threads = 8 wave32; 64x128 C tile; BK=32.
// Wave w: rows (w&3)*16, cols (w>>2)*64 => 4 accumulators of 16x16.
// A tile staged with GLOBAL_LOAD_ASYNC_TO_LDS_B64; B staged transposed via
// coalesced b128 loads + b16 LDS scatter.
// ---------------------------------------------------------------------------
#define BM 64
#define BN 128
#define BK 32
#define LDT 36   // 72B row stride: 8B-aligned rows, 16 rows hit 16 distinct banks

__global__ __launch_bounds__(256)
void k_gemm_bf16_wmma(const __bf16* __restrict__ A, const __bf16* __restrict__ B,
                      float* __restrict__ C, int M, int N, int K) {
  __shared__ __bf16 As[BM * LDT];   // [row][k]
  __shared__ __bf16 Bs[BN * LDT];   // transposed: [n][k]

  const int t    = threadIdx.x;
  const int lane = t & 31;
  const int wave = t >> 5;
  const int wr   = wave & 3;          // 16-row tile
  const int wc   = (wave >> 2) * 64;  // 64-col group (4 x 16)
  const long m0  = (long)blockIdx.y * BM;
  const long n0  = (long)blockIdx.x * BN;

  v8f acc0 = {}, acc1 = {}, acc2 = {}, acc3 = {};

  // A-fragment mapping (ISA 16-bit A 16x32): M = lane&15, kbase = 0|8
  const int am  = lane & 15;
  const int akb = (lane >> 4) << 3;
  // B-fragment mapping (32x16): N = lane&15, kbase = 0|8
  const int bn  = lane & 15;
  const int bkb = (lane >> 4) << 3;

  for (int k0 = 0; k0 < K; k0 += BK) {
    // ---- A tile via async copies: 64 rows x 64B = 512 b64 chunks ----
#pragma unroll
    for (int i = 0; i < 2; ++i) {
      int c  = i * 256 + t;
      int r  = c >> 3;          // row 0..63
      int cj = c & 7;           // 8B chunk within row
      async_copy_b64(&As[r * LDT + cj * 4],
                     &A[(m0 + r) * (long)K + k0 + cj * 4]);
    }
    // ---- B tile transposed: coalesced 16B loads, b16 scatter to LDS ----
    // 128 n x 32 k = 512 chunks of 8 bf16 along n
#pragma unroll
    for (int i = 0; i < 2; ++i) {
      int c  = i * 256 + t;
      int k  = c >> 4;           // 0..31
      int nb = (c & 15) * 8;     // n chunk base 0..120
      v8bf vv = *(const v8bf*)&B[(long)(k0 + k) * N + n0 + nb];
#pragma unroll
      for (int j = 0; j < 8; ++j)
        Bs[(nb + j) * LDT + k] = vv[j];
    }
    wait_async0();
    __syncthreads();

    // ---- fragments + 4 WMMAs (A reused across all 4) ----
    v16bf a;
    const __bf16* ap = As + (wr * 16 + am) * LDT;
#pragma unroll
    for (int q = 0; q < 8; ++q) {
      int k = ((q & 4) << 2) + akb + ((q & 3) << 1);
      a[2 * q]     = ap[k];
      a[2 * q + 1] = ap[k + 1];
    }
    v16bf b0, b1, b2, b3;
    const __bf16* bp0 = Bs + (wc + bn) * LDT;
    const __bf16* bp1 = bp0 + 16 * LDT;
    const __bf16* bp2 = bp0 + 32 * LDT;
    const __bf16* bp3 = bp0 + 48 * LDT;
#pragma unroll
    for (int q = 0; q < 8; ++q) {
      int k = ((q & 4) << 2) + bkb + ((q & 3) << 1);
      b0[2 * q] = bp0[k]; b0[2 * q + 1] = bp0[k + 1];
      b1[2 * q] = bp1[k]; b1[2 * q + 1] = bp1[k + 1];
      b2[2 * q] = bp2[k]; b2[2 * q + 1] = bp2[k + 1];
      b3[2 * q] = bp3[k]; b3[2 * q + 1] = bp3[k + 1];
    }

    acc0 = __builtin_amdgcn_wmma_f32_16x16x32_bf16(false, a, false, b0,
                                                   (short)0, acc0, false, false);
    acc1 = __builtin_amdgcn_wmma_f32_16x16x32_bf16(false, a, false, b1,
                                                   (short)0, acc1, true, false);
    acc2 = __builtin_amdgcn_wmma_f32_16x16x32_bf16(false, a, false, b2,
                                                   (short)0, acc2, true, false);
    acc3 = __builtin_amdgcn_wmma_f32_16x16x32_bf16(false, a, false, b3,
                                                   (short)0, acc3, true, false);
    __syncthreads();
  }

  // ---- store C (lane 0-15: M=r, lane 16-31: M=r+8; N = lane&15) ----
  const long crow = m0 + wr * 16 + ((lane >> 4) << 3);
  const long ccol = n0 + wc + (lane & 15);
#pragma unroll
  for (int r = 0; r < 8; ++r) {
    C[(crow + r) * N + ccol]      = acc0[r];
    C[(crow + r) * N + ccol + 16] = acc1[r];
    C[(crow + r) * N + ccol + 32] = acc2[r];
    C[(crow + r) * N + ccol + 48] = acc3[r];
  }
}

// ---------------------------------------------------------------------------
// fp32 -> bf16 conversion with zero padding
// ---------------------------------------------------------------------------
__global__ void k_cvtA(const float* __restrict__ in, __bf16* out,
                       int M, int K, int Kp) {
  long i = (long)blockIdx.x * blockDim.x + threadIdx.x;
  long s = (long)gridDim.x * blockDim.x;
  long total = (long)M * Kp;
  for (; i < total; i += s) {
    int r = (int)(i / Kp), k = (int)(i % Kp);
    out[i] = (k < K) ? f2bf(in[(long)r * K + k]) : f2bf(0.0f);
  }
}
__global__ void k_cvtB(const float* __restrict__ in, __bf16* out,
                       int K, int Kp, int N) {
  long i = (long)blockIdx.x * blockDim.x + threadIdx.x;
  long s = (long)gridDim.x * blockDim.x;
  long total = (long)Kp * N;
  for (; i < total; i += s) {
    int k = (int)(i / N);
    out[i] = (k < K) ? f2bf(in[i]) : f2bf(0.0f);
  }
}

// ---------------------------------------------------------------------------
// Graph / attention kernels
// ---------------------------------------------------------------------------
__global__ void k_fill(float* p, float v, long n) {
  long i = (long)blockIdx.x * blockDim.x + threadIdx.x;
  long s = (long)gridDim.x * blockDim.x;
  for (; i < n; i += s) p[i] = v;
}

__global__ void k_deg_loop(const int* __restrict__ ei, const float* __restrict__ ea,
                           float* deg, float* lsum, int E, int EDIM) {
  int e = blockIdx.x * blockDim.x + threadIdx.x;
  if (e >= E) return;
  int d = ei[E + e];
  atomicAdd(&deg[d], 1.0f);
  for (int j = 0; j < EDIM; ++j)
    atomicAdd(&lsum[d * EDIM + j], ea[e * EDIM + j]);
}

__global__ void k_loop_norm(const float* lsum, const float* deg, float* lattr,
                            int N, int EDIM) {
  int i = blockIdx.x * blockDim.x + threadIdx.x;
  if (i >= N * EDIM) return;
  lattr[i] = lsum[i] / fmaxf(deg[i / EDIM], 1.0f);
}

__global__ void k_reduce_c(const float* __restrict__ We, const float* __restrict__ aedge,
                           float* c, int EDIM, int H, int D) {
  int i = blockIdx.x * blockDim.x + threadIdx.x;
  if (i >= EDIM * H) return;
  int k = i / H, h = i % H;
  float s = 0.0f;
  const float* wrow = We + (long)k * H * D + (long)h * D;
  const float* ar = aedge + (long)h * D;
  for (int d = 0; d < D; ++d) s += wrow[d] * ar[d];
  c[k * H + h] = s;
}

__global__ void k_sedge(const float* __restrict__ ea, const float* __restrict__ lattr,
                        const float* __restrict__ c, float* se,
                        int E, int E2, int EDIM, int H) {
  int i = blockIdx.x * blockDim.x + threadIdx.x;
  if (i >= E2 * H) return;
  int e = i / H, h = i % H;
  const float* row = (e < E) ? (ea + (long)e * EDIM) : (lattr + (long)(e - E) * EDIM);
  float s = 0.0f;
  for (int k = 0; k < EDIM; ++k) s += row[k] * c[k * H + h];
  se[i] = s;
}

__global__ void k_node_scores(const float* __restrict__ xs,
                              const float* __restrict__ asrc,
                              const float* __restrict__ adst,
                              float* ssrc, float* sdst, int N, int H, int D) {
  int i = blockIdx.x * blockDim.x + threadIdx.x;
  if (i >= N * H) return;
  int n = i / H, h = i % H;
  const float4* xp = (const float4*)(xs + ((long)n * H + h) * D);
  const float4* sp = (const float4*)(asrc + (long)h * D);
  const float4* dp = (const float4*)(adst + (long)h * D);
  float a = 0.0f, b = 0.0f;
  for (int d = 0; d < D / 4; ++d) {
    float4 x = xp[d], s4 = sp[d], d4 = dp[d];
    a += x.x * s4.x + x.y * s4.y + x.z * s4.z + x.w * s4.w;
    b += x.x * d4.x + x.y * d4.y + x.z * d4.z + x.w * d4.w;
  }
  ssrc[i] = a; sdst[i] = b;
}

__global__ void k_logit_max(const int* __restrict__ ei, const float* __restrict__ ssrc,
                            const float* __restrict__ sdst, const float* __restrict__ se,
                            float* att, float* m, int E, int E2, int H) {
  int i = blockIdx.x * blockDim.x + threadIdx.x;
  if (i >= E2 * H) return;
  int e = i / H, h = i % H;
  int s, d;
  if (e < E) { s = ei[e]; d = ei[E + e]; } else { s = d = e - E; }
  float lg = ssrc[s * H + h] + sdst[d * H + h] + se[i];
  lg = (lg > 0.0f) ? lg : NEG_SLOPE * lg;
  att[i] = lg;
  atomicMaxF(&m[d * H + h], lg);
}

__global__ void k_exp_sum(const int* __restrict__ ei, const float* __restrict__ m,
                          float* att, float* z, int E, int E2, int H) {
  int i = blockIdx.x * blockDim.x + threadIdx.x;
  if (i >= E2 * H) return;
  int e = i / H, h = i % H;
  int d = (e < E) ? ei[E + e] : (e - E);
  float p = __expf(att[i] - m[d * H + h]);
  att[i] = p;
  atomicAdd(&z[d * H + h], p);
}

__global__ void k_alpha(const int* __restrict__ ei, const float* __restrict__ z,
                        float* att, int E, int E2, int H) {
  int i = blockIdx.x * blockDim.x + threadIdx.x;
  if (i >= E2 * H) return;
  int e = i / H, h = i % H;
  int d = (e < E) ? ei[E + e] : (e - E);
  att[i] = att[i] / (z[d * H + h] + 1e-16f);
}

__global__ void k_aggregate(const int* __restrict__ ei, const float* __restrict__ att,
                            const float* __restrict__ xs, float* out,
                            int E, int H, int D) {
  int eh = blockIdx.x;
  int e = eh / H, h = eh % H;
  int s, d;
  if (e < E) { s = ei[e]; d = ei[E + e]; } else { s = d = e - E; }
  float al = att[eh];
  const float* xp = xs + ((long)s * H + h) * D;
  float* op = out + ((long)d * H + h) * D;
  for (int j = threadIdx.x; j < D; j += blockDim.x)
    atomicAdd(&op[j], al * xp[j]);
}

__global__ void k_bias_elu(float* h, const float* __restrict__ b, long n, int cols) {
  long i = (long)blockIdx.x * blockDim.x + threadIdx.x;
  long s = (long)gridDim.x * blockDim.x;
  for (; i < n; i += s) {
    float v = h[i] + b[i % cols];
    h[i] = (v > 0.0f) ? v : (__expf(v) - 1.0f);
  }
}

__global__ void k_bias_elu_cvt(const float* __restrict__ hf, const float* __restrict__ b,
                               __bf16* hb, long n, int cols) {
  long i = (long)blockIdx.x * blockDim.x + threadIdx.x;
  long s = (long)gridDim.x * blockDim.x;
  for (; i < n; i += s) {
    float v = hf[i] + b[i % cols];
    v = (v > 0.0f) ? v : (__expf(v) - 1.0f);
    hb[i] = f2bf(v);
  }
}

__global__ void k_pool(const int* __restrict__ batch, const float* __restrict__ h,
                       float* pool, int D) {
  int n = blockIdx.x;
  int g = batch[n];
  const float* hp = h + (long)n * D;
  float* pp = pool + (long)g * D;
  for (int d = threadIdx.x; d < D; d += blockDim.x) atomicMaxF(&pp[d], hp[d]);
}

__global__ void k_write_out(const float* __restrict__ pool, const int* __restrict__ ei,
                            const float* __restrict__ alpha2, float* outp,
                            int G, int D, int E, int N) {
  int E2 = E + N;
  long total = (long)G * D + 3L * E2;
  long i = (long)blockIdx.x * blockDim.x + threadIdx.x;
  long s = (long)gridDim.x * blockDim.x;
  for (; i < total; i += s) {
    if (i < (long)G * D) {
      outp[i] = pool[i];
    } else if (i < (long)G * D + 2L * E2) {
      long j = i - (long)G * D;
      int row = (int)(j / E2);
      int e = (int)(j % E2);
      int val = (e < E) ? ei[row * E + e] : (e - E);
      ((int*)outp)[i] = val;
    } else {
      long e = i - (long)G * D - 2L * E2;
      outp[i] = alpha2[e];
    }
  }
}

// ---------------------------------------------------------------------------
// Orchestration
// ---------------------------------------------------------------------------
extern "C" void kernel_launch(void* const* d_in, const int* in_sizes, int n_in,
                              void* d_out, int out_size, void* d_ws, size_t ws_size,
                              hipStream_t stream) {
  const float* x1     = (const float*)d_in[0];
  const int*   ei     = (const int*)d_in[1];
  const float* eattr  = (const float*)d_in[2];
  const int*   batch  = (const int*)d_in[3];
  const float* W1     = (const float*)d_in[5];
  const float* We1    = (const float*)d_in[6];
  const float* asrc1  = (const float*)d_in[7];
  const float* adst1  = (const float*)d_in[8];
  const float* aedge1 = (const float*)d_in[9];
  const float* b1     = (const float*)d_in[10];
  const float* W2     = (const float*)d_in[11];
  const float* We2    = (const float*)d_in[12];
  const float* asrc2  = (const float*)d_in[13];
  const float* adst2  = (const float*)d_in[14];
  const float* aedge2 = (const float*)d_in[15];
  const float* b2     = (const float*)d_in[16];

  const int N = 8192, E = 32768, F = 93, EDIM = 11, H = 10, D = 512;
  const int HD = H * D, G = 64, E2 = E + N;
  const int Fp = 96;            // F padded to BK multiple
  const float NINF = -__builtin_inff();

  char* wsp = (char*)d_ws;
  size_t off = 0;
  auto carveB = [&](size_t bytes) -> void* {
    void* p = (void*)(wsp + off);
    off = (off + bytes + 255) & ~(size_t)255;
    return p;
  };
  auto carve = [&](size_t nelem) -> float* { return (float*)carveB(nelem * 4); };

  float*  xs1   = carve((size_t)N * HD);
  float*  h1f   = carve((size_t)N * HD);
  float*  xs2   = carve((size_t)N * D);
  float*  out2  = carve((size_t)N * D);
  __bf16* x1b   = (__bf16*)carveB((size_t)N * Fp * 2);
  __bf16* W1b   = (__bf16*)carveB((size_t)Fp * HD * 2);
  __bf16* h1b   = (__bf16*)carveB((size_t)N * HD * 2);
  __bf16* W2b   = (__bf16*)carveB((size_t)HD * D * 2);
  float*  deg   = carve(N);
  float*  lsum  = carve((size_t)N * EDIM);
  float*  lattr = carve((size_t)N * EDIM);
  float*  c1    = carve((size_t)EDIM * H);
  float*  se1   = carve((size_t)E2 * H);
  float*  ss1   = carve((size_t)N * H);
  float*  sd1   = carve((size_t)N * H);
  float*  m1    = carve((size_t)N * H);
  float*  z1    = carve((size_t)N * H);
  float*  att1  = carve((size_t)E2 * H);
  float*  c2    = carve(EDIM);
  float*  se2   = carve(E2);
  float*  ss2   = carve(N);
  float*  sd2   = carve(N);
  float*  m2    = carve(N);
  float*  z2    = carve(N);
  float*  att2  = carve(E2);
  float*  pool  = carve((size_t)G * D);

  const int TB = 256;
  auto blocks = [](long n, int tb) { return (int)((n + tb - 1) / tb); };

  // ---- init ----
  k_fill<<<1024, TB, 0, stream>>>(deg, 0.0f, N);
  k_fill<<<1024, TB, 0, stream>>>(lsum, 0.0f, (long)N * EDIM);
  k_fill<<<1024, TB, 0, stream>>>(m1, NINF, (long)N * H);
  k_fill<<<1024, TB, 0, stream>>>(z1, 0.0f, (long)N * H);
  k_fill<<<4096, TB, 0, stream>>>(h1f, 0.0f, (long)N * HD);
  k_fill<<<1024, TB, 0, stream>>>(m2, NINF, N);
  k_fill<<<1024, TB, 0, stream>>>(z2, 0.0f, N);
  k_fill<<<2048, TB, 0, stream>>>(out2, 0.0f, (long)N * D);
  k_fill<<<128, TB, 0, stream>>>(pool, NINF, (long)G * D);

  // ---- self-loop edge attrs (scatter-mean) ----
  k_deg_loop<<<blocks(E, TB), TB, 0, stream>>>(ei, eattr, deg, lsum, E, EDIM);
  k_loop_norm<<<blocks((long)N * EDIM, TB), TB, 0, stream>>>(lsum, deg, lattr, N, EDIM);

  // ---- layer 1: xs1 = x1 @ W1 (bf16 WMMA) ----
  k_cvtA<<<2048, TB, 0, stream>>>(x1, x1b, N, F, Fp);
  k_cvtB<<<2048, TB, 0, stream>>>(W1, W1b, F, Fp, HD);
  k_gemm_bf16_wmma<<<dim3(HD / BN, N / BM), 256, 0, stream>>>(x1b, W1b, xs1, N, HD, Fp);

  k_reduce_c<<<blocks(EDIM * H, TB), TB, 0, stream>>>(We1, aedge1, c1, EDIM, H, D);
  k_sedge<<<blocks((long)E2 * H, TB), TB, 0, stream>>>(eattr, lattr, c1, se1, E, E2, EDIM, H);
  k_node_scores<<<blocks((long)N * H, TB), TB, 0, stream>>>(xs1, asrc1, adst1, ss1, sd1, N, H, D);
  k_logit_max<<<blocks((long)E2 * H, TB), TB, 0, stream>>>(ei, ss1, sd1, se1, att1, m1, E, E2, H);
  k_exp_sum<<<blocks((long)E2 * H, TB), TB, 0, stream>>>(ei, m1, att1, z1, E, E2, H);
  k_alpha<<<blocks((long)E2 * H, TB), TB, 0, stream>>>(ei, z1, att1, E, E2, H);
  k_aggregate<<<E2 * H, 128, 0, stream>>>(ei, att1, xs1, h1f, E, H, D);
  k_bias_elu_cvt<<<4096, TB, 0, stream>>>(h1f, b1, h1b, (long)N * HD, HD);

  // ---- layer 2 (H=1): xs2 = elu(h1) @ W2 (bf16 WMMA) ----
  k_cvtB<<<4096, TB, 0, stream>>>(W2, W2b, HD, HD, D);
  k_gemm_bf16_wmma<<<dim3(D / BN, N / BM), 256, 0, stream>>>(h1b, W2b, xs2, N, D, HD);

  k_reduce_c<<<blocks(EDIM, TB), TB, 0, stream>>>(We2, aedge2, c2, EDIM, 1, D);
  k_sedge<<<blocks(E2, TB), TB, 0, stream>>>(eattr, lattr, c2, se2, E, E2, EDIM, 1);
  k_node_scores<<<blocks(N, TB), TB, 0, stream>>>(xs2, asrc2, adst2, ss2, sd2, N, 1, D);
  k_logit_max<<<blocks(E2, TB), TB, 0, stream>>>(ei, ss2, sd2, se2, att2, m2, E, E2, 1);
  k_exp_sum<<<blocks(E2, TB), TB, 0, stream>>>(ei, m2, att2, z2, E, E2, 1);
  k_alpha<<<blocks(E2, TB), TB, 0, stream>>>(ei, z2, att2, E, E2, 1);
  k_aggregate<<<E2, 128, 0, stream>>>(ei, att2, xs2, out2, E, 1, D);
  k_bias_elu<<<2048, TB, 0, stream>>>(out2, b2, (long)N * D, D);

  // ---- global max pool + output packing ----
  k_pool<<<N, 128, 0, stream>>>(batch, out2, pool, D);
  k_write_out<<<1024, TB, 0, stream>>>(pool, ei, att2, (float*)d_out, G, D, E, N);
}
